// AttentionHead_13400297963853
// MI455X (gfx1250) — matrix operand order repaired and checked by reading, not compiled, not run on hardware
//
#include <hip/hip_runtime.h>

// ---------------------------------------------------------------------------
// Causal single-head attention for MI455X (gfx1250, wave32, WMMA + TDM).
//   b=4, s=4096, d_model=1024, d_head=64
// Pipeline: [convert weights -> f16 transposed] -> [fused QKV proj, f16,
//   weights staged via tensor_load_to_lds double-buffer]
//   -> [flash attention, online softmax, causal-block skip]
//   -> [output projection, fp32 out]
// Matrix math: V_WMMA_F32_16X16X32_F16 (f16 in, f32 accumulate).
// ---------------------------------------------------------------------------

#define D_MODEL 1024
#define D_HEAD  64
#define BATCH   4
#define SEQ     4096

typedef __attribute__((ext_vector_type(16))) _Float16 v16h;
typedef __attribute__((ext_vector_type(8)))  _Float16 v8h;
typedef __attribute__((ext_vector_type(8)))  float    v8f;
typedef __attribute__((ext_vector_type(4)))  float    v4f;
typedef __attribute__((ext_vector_type(4)))  unsigned int v4u;
typedef __attribute__((ext_vector_type(8)))  int      v8i;
typedef __attribute__((ext_vector_type(4)))  int      v4i;

static __device__ __forceinline__ v8f wmma_f16(v16h a, v16h b, v8f c) {
  // D = A(16x32 f16) * B(32x16 f16) + C(16x16 f32)
  return __builtin_amdgcn_wmma_f32_16x16x32_f16(
      /*neg_a=*/false, a, /*neg_b=*/false, b,
      /*c_mod=*/(short)0, c, /*reuse_a=*/false, /*reuse_b=*/false);
}

// A-fragment (16x32 f16) from a row-major f16 matrix.
// Lane (hi = lane>>4, lo = lane&15): M = lo; element j -> K = hi*16+j
// p0/p1 point at K-offsets hi*16 and hi*16+8 of row `lo`.
static __device__ __forceinline__ v16h makeA(const _Float16* p0,
                                             const _Float16* p1) {
  v8h a0 = *(const v8h*)p0;
  v8h a1 = *(const v8h*)p1;
  v16h r;
#pragma unroll
  for (int j = 0; j < 8; ++j) { r[j] = a0[j]; r[j + 8] = a1[j]; }
  return r;
}

// ---------------------------------------------------------------------------
// TDM: async-load a 2-D f16 tile [rows=64][cols=32] (row stride 1024 elems)
// from global memory into LDS at byte offset lds_off. Tracked by TENSORcnt.
// D# bitfields per CDNA5 ISA ch.8: group0 = {flags, lds_addr, gaddr lo,
// gaddr hi | type=2}; group1 = {data_size=2B, tensor_dim0=32, tensor_dim1=64,
// tile_dim0=32, tile_dim1=64, tensor_dim0_stride=1024}.
// ---------------------------------------------------------------------------
static __device__ __forceinline__ void tdm_load_w_tile(const _Float16* gbase,
                                                       unsigned lds_off) {
  unsigned long long ga = (unsigned long long)(uintptr_t)gbase;
  v4u g0 = { 1u,                                   // count=1, user desc
             lds_off,                              // LDS byte address
             (unsigned)(ga & 0xFFFFFFFFu),         // global_addr[31:0]
             (unsigned)((ga >> 32) & 0x1FFFFFFu) | (2u << 30) };  // [56:32]|type=2
  v8i g1 = { (int)(1u << 16),    // data_size=1 (2 bytes)
             (int)(32u << 16),   // tensor_dim0 = 32 (lo16 at bits 79:64... d1[31:16])
             (int)(64u << 16),   // tensor_dim1 = 64 (d2[31:16])
             (int)(32u << 16),   // tile_dim0 = 32 (d3[31:16])
             64,                 // tile_dim1 = 64, tile_dim2 = 0
             1024,               // tensor_dim0_stride = 1024 elems
             0, 0 };             // stride hi, tensor_dim1_stride (unused, 2D)
  v4i z4 = { 0, 0, 0, 0 };
#if defined(__clang_major__) && (__clang_major__ >= 23)
  v8i z8 = { 0, 0, 0, 0, 0, 0, 0, 0 };
  __builtin_amdgcn_tensor_load_to_lds(g0, g1, z4, z4, z8, 0);
#else
  __builtin_amdgcn_tensor_load_to_lds(g0, g1, z4, z4, 0);
#endif
}

// ---------------------------------------------------------------------------
// Kernel 0: convert weights to f16, transposed for contiguous B-fragments.
//   Wqt/Wkt/Wvt: [64][1024]  (Wqt[h][d] = Wq[d][h])
//   Wot:         [1024][64]  (Wot[n][h] = Wo[h][n])
// ---------------------------------------------------------------------------
__global__ void convert_weights_kernel(const float* __restrict__ Wq,
                                       const float* __restrict__ Wk,
                                       const float* __restrict__ Wv,
                                       const float* __restrict__ Wo,
                                       _Float16* __restrict__ Wqt,
                                       _Float16* __restrict__ Wkt,
                                       _Float16* __restrict__ Wvt,
                                       _Float16* __restrict__ Wot) {
  int i = blockIdx.x * blockDim.x + threadIdx.x;
  if (i >= D_MODEL * D_HEAD) return;
  int d = i >> 6;       // 0..1023
  int h = i & 63;       // 0..63
  Wqt[(size_t)h * D_MODEL + d] = (_Float16)Wq[(size_t)d * D_HEAD + h];
  Wkt[(size_t)h * D_MODEL + d] = (_Float16)Wk[(size_t)d * D_HEAD + h];
  Wvt[(size_t)h * D_MODEL + d] = (_Float16)Wv[(size_t)d * D_HEAD + h];
  Wot[(size_t)d * D_HEAD  + h] = (_Float16)Wo[(size_t)h * D_MODEL + d];
}

// ---------------------------------------------------------------------------
// Kernel 1: fused QKV projection.
//   Q[b][s][64] (f16, pre-scaled by 1/sqrt(64)), K[b][s][64] (f16),
//   Vt[b][64][s] (f16, transposed for the P@V B-fragments).
// One wave per 16-row tile; weight tiles (3 x 64x32 f16 = 12 KB per k-step)
// staged into double-buffered LDS by wave 0 via TDM, shared by all 4 waves.
// ---------------------------------------------------------------------------
__global__ void qkv_proj_kernel(const float* __restrict__ x,
                                const _Float16* __restrict__ Wqt,
                                const _Float16* __restrict__ Wkt,
                                const _Float16* __restrict__ Wvt,
                                _Float16* __restrict__ Q,
                                _Float16* __restrict__ K,
                                _Float16* __restrict__ Vt) {
  // Double-buffered weight tiles: [parity][matrix][64 rows][32 cols] f16
  __shared__ __align__(64) _Float16 wbuf[2][3][64 * 32];

  const int lane = threadIdx.x & 31;
  const int wave = threadIdx.x >> 5;
  const int hi = lane >> 4, lo = lane & 15;
  const int tile = blockIdx.x * 4 + wave;      // 0..1023
  const int b = tile >> 8;
  const int s0 = (tile & 255) << 4;

  v8f qacc[4] = {}, kacc[4] = {}, vacc[4] = {};

  const float* xrow = x + (size_t)(b * SEQ + s0 + lo) * D_MODEL;

  // Prologue: stage k-window 0 into buffer 0.
  if (wave == 0) {
    tdm_load_w_tile(Wqt, (unsigned)(uintptr_t)&wbuf[0][0][0]);
    tdm_load_w_tile(Wkt, (unsigned)(uintptr_t)&wbuf[0][1][0]);
    tdm_load_w_tile(Wvt, (unsigned)(uintptr_t)&wbuf[0][2][0]);
  }

  int p = 0;
  for (int kk = 0; kk < D_MODEL; kk += 32) {
    // Issue next k-window into the other buffer, then wait for current one.
    // TDM ops from one wave complete in order: after issuing 3 more, waiting
    // TENSORcnt<=3 guarantees the current buffer's 3 transfers are done.
    if (wave == 0) {
      if (kk + 32 < D_MODEL) {
        const int np = p ^ 1;
        tdm_load_w_tile(Wqt + kk + 32, (unsigned)(uintptr_t)&wbuf[np][0][0]);
        tdm_load_w_tile(Wkt + kk + 32, (unsigned)(uintptr_t)&wbuf[np][1][0]);
        tdm_load_w_tile(Wvt + kk + 32, (unsigned)(uintptr_t)&wbuf[np][2][0]);
        __builtin_amdgcn_s_wait_tensorcnt(3);
      } else {
        __builtin_amdgcn_s_wait_tensorcnt(0);
      }
    }
    __syncthreads();  // publish wbuf[p] to all waves

    // A-fragment: 16 fp32 -> f16 per lane (vmem overlaps with ds/wmma)
    const float* p0 = xrow + kk + hi * 16;
    const float* p1 = xrow + kk + hi * 16 + 8;
    v4f x0 = *(const v4f*)(p0);
    v4f x1 = *(const v4f*)(p0 + 4);
    v4f x2 = *(const v4f*)(p1);
    v4f x3 = *(const v4f*)(p1 + 4);
    v16h a;
#pragma unroll
    for (int j = 0; j < 4; ++j) {
      a[j]      = (_Float16)x0[j];
      a[4 + j]  = (_Float16)x1[j];
      a[8 + j]  = (_Float16)x2[j];
      a[12 + j] = (_Float16)x3[j];
    }
#pragma unroll
    for (int t = 0; t < 4; ++t) {
      const int boff = (t * 16 + lo) * 32 + hi * 16;  // B: N=lo, K=hi*16+j
      v16h bq = *(const v16h*)(&wbuf[p][0][boff]);
      v16h bk = *(const v16h*)(&wbuf[p][1][boff]);
      v16h bv = *(const v16h*)(&wbuf[p][2][boff]);
      qacc[t] = wmma_f16(a, bq, qacc[t]);
      kacc[t] = wmma_f16(a, bk, kacc[t]);
      vacc[t] = wmma_f16(a, bv, vacc[t]);
    }
    __syncthreads();  // all waves done with wbuf[p] before it is overwritten
    p ^= 1;
  }

  const float qscale = 0.125f;  // 1/sqrt(D_HEAD)
#pragma unroll
  for (int t = 0; t < 4; ++t) {
#pragma unroll
    for (int r = 0; r < 8; ++r) {
      const int srow = s0 + r + 8 * hi;
      const int h = t * 16 + lo;
      const size_t qi = (size_t)(b * SEQ + srow) * D_HEAD + h;
      Q[qi] = (_Float16)(qacc[t][r] * qscale);
      K[qi] = (_Float16)(kacc[t][r]);
      Vt[(size_t)(b * D_HEAD + h) * SEQ + srow] = (_Float16)(vacc[t][r]);
    }
  }
}

// ---------------------------------------------------------------------------
// Kernel 2: flash attention with online softmax, causal-block skipping.
// One wave per 16 query rows; KV blocks of 32 keys. K/V stay L2-resident
// (1 MB per batch in f16), so per-wave global loads are cheap.
// ---------------------------------------------------------------------------
__global__ void flash_attn_kernel(const _Float16* __restrict__ Q,
                                  const _Float16* __restrict__ K,
                                  const _Float16* __restrict__ Vt,
                                  _Float16* __restrict__ O) {
  __shared__ __align__(16) _Float16 pbuf[4][16][32];  // per-wave P staging

  const int lane = threadIdx.x & 31;
  const int wave = threadIdx.x >> 5;
  const int hi = lane >> 4, lo = lane & 15;
  const int tile = blockIdx.x * 4 + wave;      // 0..1023
  const int b = tile >> 8;
  const int q0 = (tile & 255) << 4;

  const _Float16* Qb = Q + (size_t)b * SEQ * D_HEAD;
  const _Float16* Kb = K + (size_t)b * SEQ * D_HEAD;
  const _Float16* Vb = Vt + (size_t)b * D_HEAD * SEQ;

  v8f oacc[4] = {};
  float mrow[8], lrow[8];
#pragma unroll
  for (int r = 0; r < 8; ++r) { mrow[r] = -__builtin_inff(); lrow[r] = 0.0f; }

  // Q A-fragments (K-dim = 64 -> two k=32 windows), loaded once, reused.
  v16h aq[2];
  const _Float16* qrow = Qb + (size_t)(q0 + lo) * D_HEAD;
#pragma unroll
  for (int kk = 0; kk < 2; ++kk)
    aq[kk] = makeA(qrow + kk * 32 + hi * 16, qrow + kk * 32 + hi * 16 + 8);

  const int kv_end = q0 + 16;  // last needed key = q0+15 (causal)
  for (int kv0 = 0; kv0 < kv_end; kv0 += 32) {
    // Prefetch next KV block into cache
    __builtin_prefetch(Kb + (size_t)(kv0 + 32 + lo) * D_HEAD, 0, 1);
    __builtin_prefetch(Vb + (size_t)lo * SEQ + kv0 + 32, 0, 1);

    // ---- S = Q * K^T : two 16x16 tiles (keys kv0.. and kv0+16..) ----
    v8f s0 = {}, s1 = {};
#pragma unroll
    for (int kk = 0; kk < 2; ++kk) {
      v16h b0 = *(const v16h*)(Kb + (size_t)(kv0 + lo) * D_HEAD + kk * 32 + hi * 16);
      v16h b1 = *(const v16h*)(Kb + (size_t)(kv0 + 16 + lo) * D_HEAD + kk * 32 + hi * 16);
      s0 = wmma_f16(aq[kk], b0, s0);
      s1 = wmma_f16(aq[kk], b1, s1);
    }

    // ---- causal mask (only on diagonal blocks) ----
    if (kv0 + 31 > q0) {
#pragma unroll
      for (int r = 0; r < 8; ++r) {
        const int row = q0 + r + 8 * hi;
        if (kv0 + lo > row)      s0[r] = -__builtin_inff();
        if (kv0 + 16 + lo > row) s1[r] = -__builtin_inff();
      }
    }

    // ---- online softmax: row stats via 16-lane butterfly reductions ----
    float pr0[8], pr1[8];
#pragma unroll
    for (int r = 0; r < 8; ++r) {
      float mx = fmaxf(s0[r], s1[r]);
#pragma unroll
      for (int off = 1; off < 16; off <<= 1)
        mx = fmaxf(mx, __shfl_xor(mx, off, 32));
      const float mnew  = fmaxf(mrow[r], mx);
      const float alpha = __expf(mrow[r] - mnew);
      mrow[r] = mnew;
      const float p0 = __expf(s0[r] - mnew);
      const float p1 = __expf(s1[r] - mnew);
      pr0[r] = p0; pr1[r] = p1;
      float sum = p0 + p1;
#pragma unroll
      for (int off = 1; off < 16; off <<= 1)
        sum += __shfl_xor(sum, off, 32);
      lrow[r] = lrow[r] * alpha + sum;
#pragma unroll
      for (int t = 0; t < 4; ++t) oacc[t][r] *= alpha;
    }

    // ---- re-layout P: C-fragment -> A-fragment via per-wave LDS ----
#pragma unroll
    for (int r = 0; r < 8; ++r) {
      pbuf[wave][r + 8 * hi][lo]      = (_Float16)pr0[r];
      pbuf[wave][r + 8 * hi][16 + lo] = (_Float16)pr1[r];
    }
    const _Float16* pl = &pbuf[wave][lo][0];
    v16h pa = makeA(pl + hi * 16, pl + hi * 16 + 8);

    // ---- O += P * V ----
#pragma unroll
    for (int t = 0; t < 4; ++t) {
      v16h bv = *(const v16h*)(Vb + (size_t)(t * 16 + lo) * SEQ + kv0 + hi * 16);
      oacc[t] = wmma_f16(pa, bv, oacc[t]);
    }
  }

  // ---- epilogue: normalize and store O as f16 ----
#pragma unroll
  for (int r = 0; r < 8; ++r) {
    const float inv = 1.0f / lrow[r];
    const size_t rowoff = (size_t)(b * SEQ + q0 + r + 8 * hi) * D_HEAD;
#pragma unroll
    for (int t = 0; t < 4; ++t)
      O[rowoff + t * 16 + lo] = (_Float16)(oacc[t][r] * inv);
  }
}

// ---------------------------------------------------------------------------
// Kernel 3: output projection  out = O(f16) * Wo  -> fp32.
// One wave computes a 16x64 tile of the [16384 x 1024] result (K = 64).
// ---------------------------------------------------------------------------
__global__ void out_proj_kernel(const _Float16* __restrict__ O,
                                const _Float16* __restrict__ Wot,
                                float* __restrict__ out) {
  const int lane = threadIdx.x & 31;
  const int wave = threadIdx.x >> 5;
  const int hi = lane >> 4, lo = lane & 15;
  const int w = blockIdx.x * 4 + wave;  // 0..16383
  const int mt = w >> 4;                // 0..1023 row-tile
  const int ng = w & 15;                // 0..15 col-group (64 cols)
  const int b = mt >> 8;
  const int s0 = (mt & 255) << 4;
  const int col0 = ng * 64;

  const _Float16* orow = O + (size_t)(b * SEQ + s0 + lo) * D_HEAD;
  v16h a[2];
#pragma unroll
  for (int kk = 0; kk < 2; ++kk)
    a[kk] = makeA(orow + kk * 32 + hi * 16, orow + kk * 32 + hi * 16 + 8);

  v8f acc[4] = {};
#pragma unroll
  for (int kk = 0; kk < 2; ++kk) {
#pragma unroll
    for (int t = 0; t < 4; ++t) {
      v16h bw = *(const v16h*)(Wot + (size_t)(col0 + t * 16 + lo) * D_HEAD +
                               kk * 32 + hi * 16);
      acc[t] = wmma_f16(a[kk], bw, acc[t]);
    }
  }

#pragma unroll
  for (int r = 0; r < 8; ++r) {
    const size_t rowoff = (size_t)(b * SEQ + s0 + r + 8 * hi) * D_MODEL;
#pragma unroll
    for (int t = 0; t < 4; ++t)
      out[rowoff + col0 + t * 16 + lo] = acc[t][r];
  }
}

// ---------------------------------------------------------------------------
// Launch. Inputs: x, W_q, W_k, W_v, W_o (all fp32). Output: fp32.
// Workspace layout (f16 elems): Wqt|Wkt|Wvt (64k each) | Wot (64k)
//   | Q (1M) | K (1M) | Vt (1M) | O (1M)  => ~8.5 MB.
// ---------------------------------------------------------------------------
extern "C" void kernel_launch(void* const* d_in, const int* in_sizes, int n_in,
                              void* d_out, int out_size, void* d_ws,
                              size_t ws_size, hipStream_t stream) {
  (void)in_sizes; (void)n_in; (void)out_size; (void)ws_size;
  const float* x  = (const float*)d_in[0];
  const float* Wq = (const float*)d_in[1];
  const float* Wk = (const float*)d_in[2];
  const float* Wv = (const float*)d_in[3];
  const float* Wo = (const float*)d_in[4];

  _Float16* ws  = (_Float16*)d_ws;
  const size_t WSZ = (size_t)D_MODEL * D_HEAD;          // 65536
  const size_t TSZ = (size_t)BATCH * SEQ * D_HEAD;      // 1048576
  _Float16* Wqt = ws;
  _Float16* Wkt = Wqt + WSZ;
  _Float16* Wvt = Wkt + WSZ;
  _Float16* Wot = Wvt + WSZ;
  _Float16* Qb  = Wot + WSZ;
  _Float16* Kb  = Qb + TSZ;
  _Float16* Vtb = Kb + TSZ;
  _Float16* Ob  = Vtb + TSZ;

  convert_weights_kernel<<<(D_MODEL * D_HEAD + 255) / 256, 256, 0, stream>>>(
      Wq, Wk, Wv, Wo, Wqt, Wkt, Wvt, Wot);

  // 1024 row-tiles, 4 waves (128 threads) per block
  qkv_proj_kernel<<<256, 128, 0, stream>>>(x, Wqt, Wkt, Wvt, Qb, Kb, Vtb);

  flash_attn_kernel<<<256, 128, 0, stream>>>(Qb, Kb, Vtb, Ob);

  // 16384 wave-tiles of 16x64 -> 4096 blocks of 128 threads
  out_proj_kernel<<<4096, 128, 0, stream>>>(Ob, Wot, (float*)d_out);
}
